// CausalSelfAttention_76957224010310
// MI455X (gfx1250) — matrix operand order repaired and checked by reading, not compiled, not run on hardware
//
#include <hip/hip_runtime.h>
#include <hip/hip_bf16.h>
#include <stdint.h>

// Problem constants (match reference: B,T,C,H = 8,1024,1024,16)
#define BATCH 8
#define SEQ   1024
#define EMB   1024
#define NH    16
#define HD    64
#define QKV_N (3 * EMB)

typedef __attribute__((ext_vector_type(16))) __bf16       bf16x16;
typedef __attribute__((ext_vector_type(8)))  float        f32x8;
typedef __attribute__((ext_vector_type(4)))  unsigned int u32x4;
typedef __attribute__((ext_vector_type(4)))  int          i32x4;
typedef __attribute__((ext_vector_type(8)))  int          i32x8;

#if defined(__AMDGCN__) && __has_builtin(__builtin_amdgcn_tensor_load_to_lds)
#define HAVE_TDM 1
#else
#define HAVE_TDM 0
#endif

union Frag {           // one WMMA 16-bit A/B operand: 16 bf16 = 8 VGPRs
  bf16x16 bf;
  u32x4   q[2];
};

__device__ __forceinline__ unsigned short f32_to_bf16_bits(float f) {
  unsigned u = __float_as_uint(f);
  unsigned r = u + 0x7FFFu + ((u >> 16) & 1u);   // round-to-nearest-even
  return (unsigned short)(r >> 16);
}

__device__ __forceinline__ f32x8 wmma_bf16(const Frag& a, const Frag& b, f32x8 c) {
  return __builtin_amdgcn_wmma_f32_16x16x32_bf16(false, a.bf, false, b.bf,
                                                 (short)0, c, false, false);
}

// ---------------------------------------------------------------------------
// TDM: async-load a 2D bf16 tile [tile_d1 rows x 64 cols] into LDS, padding
// each 128B row with 16B -> LDS row stride 144B (72 halves), per D# spec §8.
__device__ __forceinline__ void tdm_load_tile(const unsigned short* gptr,
                                              unsigned lds_byte_addr,
                                              unsigned tensor_d0, unsigned tensor_d1,
                                              unsigned stride0,
                                              unsigned tile_d0, unsigned tile_d1) {
#if HAVE_TDM
  unsigned long long ga = (unsigned long long)(uintptr_t)gptr;
  u32x4 g0 = {0u, 0u, 0u, 0u};
  g0[0] = 1u;                                   // count=1 valid descriptor
  g0[1] = lds_byte_addr;                        // lds_addr
  g0[2] = (unsigned)(ga & 0xFFFFFFFFu);         // global_addr[31:0]
  g0[3] = (unsigned)((ga >> 32) & 0x1FFFFFFu)   // global_addr[56:32]
          | (2u << 30);                         // type=2 ("image")
  i32x8 g1 = {0, 0, 0, 0, 0, 0, 0, 0};
  // data_size=1 (2B) | pad_enable | pad_interval=4 (32 DW) | pad_amount=3 (4 DW)
  g1[0] = (int)((1u << 16) | (1u << 20) | (4u << 22) | (3u << 25));
  g1[1] = (int)((tensor_d0 & 0xFFFFu) << 16);                       // dim0[15:0]
  g1[2] = (int)(((tensor_d0 >> 16) & 0xFFFFu) |
                ((tensor_d1 & 0xFFFFu) << 16));                      // dim0 hi | dim1 lo
  g1[3] = (int)(((tensor_d1 >> 16) & 0xFFFFu) | (tile_d0 << 16));    // dim1 hi | tile0
  g1[4] = (int)(tile_d1 & 0xFFFFu);                                  // tile1 (tile2=0)
  g1[5] = (int)stride0;                                              // dim0_stride lo
  i32x4 gz4 = {0, 0, 0, 0};                     // 2-D tile: dims 2..4 unused
#if __clang_major__ >= 23
  i32x8 gz8 = {0, 0, 0, 0, 0, 0, 0, 0};
  __builtin_amdgcn_tensor_load_to_lds(g0, g1, gz4, gz4, gz8, 0);
#else
  __builtin_amdgcn_tensor_load_to_lds(g0, g1, gz4, gz4, 0);
#endif
#else
  (void)gptr; (void)lds_byte_addr; (void)tensor_d0; (void)tensor_d1;
  (void)stride0; (void)tile_d0; (void)tile_d1;
#endif
}

// ---------------------------------------------------------------------------
// f32 -> bf16 elementwise
__global__ void __launch_bounds__(256) k_cvt_bf16(const float* __restrict__ in,
                                                  unsigned short* __restrict__ out,
                                                  int n) {
  int i = blockIdx.x * 256 + threadIdx.x;
  if (i < n) out[i] = f32_to_bf16_bits(in[i]);
}

// W [K][N] f32 row-major -> Wt [N][K] bf16 (GEMM B-fragments become contiguous)
__global__ void __launch_bounds__(256) k_cvt_transpose(const float* __restrict__ W,
                                                       unsigned short* __restrict__ Wt,
                                                       int K, int N) {
  int idx = blockIdx.x * 256 + threadIdx.x;
  if (idx >= K * N) return;
  int nrow = idx / K;
  int kk   = idx - nrow * K;
  Wt[idx] = f32_to_bf16_bits(W[(size_t)kk * N + nrow]);
}

// ---------------------------------------------------------------------------
// GEMM core: 128x128 block = A[M x K] * Bt[N x K]^T, bf16 in, f32 acc.
// 8 waves, wave grid 4(M) x 2(N); each wave owns 32x64 = 2x4 WMMA tiles.
// Tiles staged via TDM (double-buffered, TENSORcnt) with 144B padded rows.
#define LDS_STRIDE 72                   // halves per row (64 data + 8 pad)
#define ABUF_H (128 * LDS_STRIDE)       // halves per A/B buffer
#define ABUF_BYTES (ABUF_H * 2)

__device__ __forceinline__ void gemm_core(const unsigned short* __restrict__ A,
                                          const unsigned short* __restrict__ Bt,
                                          int K, int Mtot, int Ntot,
                                          unsigned short* ldsA,   // [2*128*72]
                                          unsigned short* ldsB,   // [2*128*72]
                                          f32x8 acc[2][4]) {
  const int tid  = threadIdx.x;
  const int lane = tid & 31, l16 = lane & 15, kg = lane >> 4;
  const int w = tid >> 5, wm = w >> 1, wn = w & 1;
  const size_t m0 = (size_t)blockIdx.y * 128;
  const size_t n0 = (size_t)blockIdx.x * 128;
  const int NK = K >> 6;                         // 64-wide K tiles

#if HAVE_TDM
  const unsigned ldsAb = (unsigned)(uintptr_t)(void*)ldsA;
  const unsigned ldsBb = (unsigned)(uintptr_t)(void*)ldsB;
  if (w == 0) {                                  // one wave drives the TDM
    tdm_load_tile(A + m0 * K, ldsAb, K, (unsigned)Mtot, (unsigned)K, 64, 128);
    tdm_load_tile(Bt + n0 * K, ldsBb, K, (unsigned)Ntot, (unsigned)K, 64, 128);
  }
#endif

  for (int kt = 0; kt < NK; ++kt) {
    const int cur = kt & 1;
    const int k0 = kt << 6;
#if HAVE_TDM
    if (w == 0) {
      if (kt + 1 < NK) {                         // prefetch next tile pair
        const int nxt = cur ^ 1;
        tdm_load_tile(A + m0 * K + (k0 + 64), ldsAb + nxt * ABUF_BYTES,
                      K, (unsigned)Mtot, (unsigned)K, 64, 128);
        tdm_load_tile(Bt + n0 * K + (k0 + 64), ldsBb + nxt * ABUF_BYTES,
                      K, (unsigned)Ntot, (unsigned)K, 64, 128);
        __builtin_amdgcn_s_wait_tensorcnt(2);    // current pair resident
      } else {
        __builtin_amdgcn_s_wait_tensorcnt(0);
      }
    }
#else
    // Fallback: manual staging into buffer 0 (cur forced to 0 below)
#pragma unroll
    for (int i = 0; i < 4; ++i) {
      int chunk = tid + i * 256;                 // A: 128 rows x 8 chunks
      int row = chunk >> 3, c = chunk & 7;
      *(u32x4*)(ldsA + row * LDS_STRIDE + c * 8) =
          *(const u32x4*)(A + (m0 + row) * K + k0 + c * 8);
    }
#pragma unroll
    for (int i = 0; i < 4; ++i) {
      int chunk = tid + i * 256;                 // B: 128 rows x 8 chunks
      int row = chunk >> 3, c = chunk & 7;
      *(u32x4*)(ldsB + row * LDS_STRIDE + c * 8) =
          *(const u32x4*)(Bt + (n0 + row) * K + k0 + c * 8);
    }
#endif
    __syncthreads();

#if HAVE_TDM
    const unsigned short* bufA = ldsA + cur * ABUF_H;
    const unsigned short* bufB = ldsB + cur * ABUF_H;
#else
    const unsigned short* bufA = ldsA;
    const unsigned short* bufB = ldsB;
#endif
#pragma unroll
    for (int ks = 0; ks < 64; ks += 32) {
      Frag af[2], bfr[4];
      // A-frag: halves0-7 = K kg*8.. ; halves8-15 = 16+kg*8..
#pragma unroll
      for (int ms = 0; ms < 2; ++ms) {
        const unsigned short* p = bufA + (wm * 32 + ms * 16 + l16) * LDS_STRIDE + ks;
        af[ms].q[0] = *(const u32x4*)(p + kg * 8);
        af[ms].q[1] = *(const u32x4*)(p + 16 + kg * 8);
      }
      // B-frag: halves0-7 = K kg*8.. ; halves8-15 = 16+kg*8..
#pragma unroll
      for (int ns = 0; ns < 4; ++ns) {
        const unsigned short* p =
            bufB + (wn * 64 + ns * 16 + l16) * LDS_STRIDE + ks + kg * 8;
        bfr[ns].q[0] = *(const u32x4*)(p);
        bfr[ns].q[1] = *(const u32x4*)(p + 16);
      }
#pragma unroll
      for (int ms = 0; ms < 2; ++ms)
#pragma unroll
        for (int ns = 0; ns < 4; ++ns)
          acc[ms][ns] = wmma_bf16(af[ms], bfr[ns], acc[ms][ns]);
    }
    __syncthreads();   // all waves done reading before buffer is overwritten
  }
}

// ---------------------------------------------------------------------------
// QKV GEMM: qkv = x @ Wqkv + bqkv ; scatter Q/K [B,H,T,hd] bf16, V^T [B,H,hd,T]
__global__ void __launch_bounds__(256) k_gemm_qkv(const unsigned short* __restrict__ A,
                                                  const unsigned short* __restrict__ Bt,
                                                  const float* __restrict__ bias,
                                                  unsigned short* __restrict__ Qb,
                                                  unsigned short* __restrict__ Kb,
                                                  unsigned short* __restrict__ Vt) {
  __shared__ unsigned short ldsA[2 * ABUF_H];
  __shared__ unsigned short ldsB[2 * ABUF_H];
  f32x8 acc[2][4] = {};
  gemm_core(A, Bt, EMB, BATCH * SEQ, QKV_N, ldsA, ldsB, acc);

  const int lane = threadIdx.x & 31, l16 = lane & 15, kg = lane >> 4;
  const int w = threadIdx.x >> 5, wm = w >> 1, wn = w & 1;
  const int gm_base = blockIdx.y * 128 + wm * 32;
  const int gn_base = blockIdx.x * 128 + wn * 64;
#pragma unroll
  for (int ms = 0; ms < 2; ++ms)
#pragma unroll
    for (int ns = 0; ns < 4; ++ns) {
      const int n = gn_base + ns * 16 + l16;     // [0, 3072)
      const float bv = bias[n];
      const int which = n >> 10;                  // 0=Q 1=K 2=V
      const int c = n & (EMB - 1);
      const int hh = c >> 6, dd = c & (HD - 1);
#pragma unroll
      for (int r = 0; r < 8; ++r) {
        const int m = gm_base + ms * 16 + r + 8 * kg;
        const int bb = m >> 10, tt = m & (SEQ - 1);
        const unsigned short hv = f32_to_bf16_bits(acc[ms][ns][r] + bv);
        if (which == 0)
          Qb[(((size_t)(bb * NH + hh)) * SEQ + tt) * HD + dd] = hv;
        else if (which == 1)
          Kb[(((size_t)(bb * NH + hh)) * SEQ + tt) * HD + dd] = hv;
        else
          Vt[(((size_t)(bb * NH + hh)) * HD + dd) * SEQ + tt] = hv;
      }
    }
}

// Proj GEMM: out = y @ Wproj + bproj, fp32 output
__global__ void __launch_bounds__(256) k_gemm_proj(const unsigned short* __restrict__ A,
                                                   const unsigned short* __restrict__ Bt,
                                                   const float* __restrict__ bias,
                                                   float* __restrict__ out) {
  __shared__ unsigned short ldsA[2 * ABUF_H];
  __shared__ unsigned short ldsB[2 * ABUF_H];
  f32x8 acc[2][4] = {};
  gemm_core(A, Bt, EMB, BATCH * SEQ, EMB, ldsA, ldsB, acc);

  const int lane = threadIdx.x & 31, l16 = lane & 15, kg = lane >> 4;
  const int w = threadIdx.x >> 5, wm = w >> 1, wn = w & 1;
  const int gm_base = blockIdx.y * 128 + wm * 32;
  const int gn_base = blockIdx.x * 128 + wn * 64;
#pragma unroll
  for (int ms = 0; ms < 2; ++ms)
#pragma unroll
    for (int ns = 0; ns < 4; ++ns) {
      const int n = gn_base + ns * 16 + l16;
      const float bv = bias[n];
#pragma unroll
      for (int r = 0; r < 8; ++r) {
        const int m = gm_base + ms * 16 + r + 8 * kg;
        out[(size_t)m * EMB + n] = acc[ms][ns][r] + bv;
      }
    }
}

// ---------------------------------------------------------------------------
// Flash attention: 1 wave / 16-query tile / head. Online softmax in f32.
// Q,K: [B,H,T,64] bf16 ; Vt: [B,H,64,T] bf16 ; Y: [B*T, C] bf16
__global__ void __launch_bounds__(32) k_attn(const unsigned short* __restrict__ Qb,
                                             const unsigned short* __restrict__ Kb,
                                             const unsigned short* __restrict__ Vt,
                                             unsigned short* __restrict__ Y) {
  __shared__ unsigned short ldsP[16 * 40];  // 16x32 P tile, rows padded to 40
  const int lane = threadIdx.x & 31, l16 = lane & 15, kg = lane >> 4;
  const int q0 = blockIdx.x * 16;
  const int h = blockIdx.y, b = blockIdx.z;
  const unsigned short* Qh = Qb + ((size_t)(b * NH + h) * SEQ) * HD;
  const unsigned short* Kh = Kb + ((size_t)(b * NH + h) * SEQ) * HD;
  const unsigned short* Vh = Vt + ((size_t)(b * NH + h) * HD) * SEQ;

  Frag qf[2];                    // resident Q fragments (d 0..31, 32..63)
  {
    const unsigned short* p = Qh + (size_t)(q0 + l16) * HD;
#pragma unroll
    for (int s = 0; s < 2; ++s) {
      qf[s].q[0] = *(const u32x4*)(p + s * 32 + kg * 8);
      qf[s].q[1] = *(const u32x4*)(p + s * 32 + 16 + kg * 8);
    }
  }

  f32x8 acc[4] = {};             // O tile 16 x 64
  float mstate[8], lstate[8];    // lane owns rows r + 8*kg
#pragma unroll
  for (int r = 0; r < 8; ++r) { mstate[r] = -1e30f; lstate[r] = 0.f; }

  const int nblk = (q0 + 47) >> 5;
  for (int j = 0; j < nblk; ++j) {
    const int j0 = j * 32;
    f32x8 s[2];
#pragma unroll
    for (int n = 0; n < 2; ++n) {
      Frag kf0, kf1;
      const unsigned short* p = Kh + (size_t)(j0 + n * 16 + l16) * HD + kg * 8;
      kf0.q[0] = *(const u32x4*)(p);      kf0.q[1] = *(const u32x4*)(p + 16);
      kf1.q[0] = *(const u32x4*)(p + 32); kf1.q[1] = *(const u32x4*)(p + 48);
      f32x8 c = {};
      c = wmma_bf16(qf[0], kf0, c);
      c = wmma_bf16(qf[1], kf1, c);
      s[n] = c;
    }

    float ascale[8];
#pragma unroll
    for (int r = 0; r < 8; ++r) {
      const int query = q0 + r + 8 * kg;
      float v0 = s[0][r] * 0.125f;               // 1/sqrt(64)
      float v1 = s[1][r] * 0.125f;
      v0 = (j0 + l16 <= query) ? v0 : -1e30f;    // branch-free causal mask
      v1 = (j0 + 16 + l16 <= query) ? v1 : -1e30f;
      float rmax = fmaxf(v0, v1);
      rmax = fmaxf(rmax, __shfl_xor(rmax, 1));
      rmax = fmaxf(rmax, __shfl_xor(rmax, 2));
      rmax = fmaxf(rmax, __shfl_xor(rmax, 4));
      rmax = fmaxf(rmax, __shfl_xor(rmax, 8));
      const float mnew = fmaxf(mstate[r], rmax);
      const float alpha = __expf(mstate[r] - mnew);
      const float p0 = __expf(v0 - mnew);
      const float p1 = __expf(v1 - mnew);
      float ps = p0 + p1;
      ps += __shfl_xor(ps, 1);
      ps += __shfl_xor(ps, 2);
      ps += __shfl_xor(ps, 4);
      ps += __shfl_xor(ps, 8);
      lstate[r] = lstate[r] * alpha + ps;
      mstate[r] = mnew;
      ascale[r] = alpha;
      const int prow = r + 8 * kg;               // C/D layout -> LDS
      ldsP[prow * 40 + l16]      = f32_to_bf16_bits(p0);
      ldsP[prow * 40 + 16 + l16] = f32_to_bf16_bits(p1);
    }
#pragma unroll
    for (int nd = 0; nd < 4; ++nd)
#pragma unroll
      for (int r = 0; r < 8; ++r) acc[nd][r] *= ascale[r];

    __syncthreads();

    Frag pf;                                     // P as A-fragment (k = 32 keys)
    {
      const unsigned short* p = ldsP + l16 * 40;
      pf.q[0] = *(const u32x4*)(p + kg * 8);
      pf.q[1] = *(const u32x4*)(p + 16 + kg * 8);
    }
#pragma unroll
    for (int nd = 0; nd < 4; ++nd) {
      Frag vf;
      const unsigned short* p = Vh + (size_t)(nd * 16 + l16) * SEQ + j0 + kg * 8;
      vf.q[0] = *(const u32x4*)(p);
      vf.q[1] = *(const u32x4*)(p + 16);
      acc[nd] = wmma_bf16(pf, vf, acc[nd]);
    }
    __syncthreads();
  }

#pragma unroll
  for (int r = 0; r < 8; ++r) {
    const float inv = 1.0f / lstate[r];
    const int row = q0 + r + 8 * kg;
#pragma unroll
    for (int nd = 0; nd < 4; ++nd)
      Y[((size_t)(b * SEQ + row)) * EMB + h * HD + nd * 16 + l16] =
          f32_to_bf16_bits(acc[nd][r] * inv);
  }
}

// ---------------------------------------------------------------------------
extern "C" void kernel_launch(void* const* d_in, const int* in_sizes, int n_in,
                              void* d_out, int out_size, void* d_ws, size_t ws_size,
                              hipStream_t stream) {
  (void)in_sizes; (void)n_in; (void)out_size; (void)ws_size;
  const float* x     = (const float*)d_in[0];
  const float* Wqkv  = (const float*)d_in[1];
  const float* bqkv  = (const float*)d_in[2];
  const float* Wproj = (const float*)d_in[3];
  const float* bproj = (const float*)d_in[4];
  float* out = (float*)d_out;

  // Workspace layout (~88 MiB total), 256B-aligned slices
  char* ws = (char*)d_ws;
  size_t off = 0;
  auto alloc = [&](size_t bytes) {
    char* p = ws + off;
    off += (bytes + 255) & ~(size_t)255;
    return p;
  };
  unsigned short* xb  = (unsigned short*)alloc((size_t)BATCH * SEQ * EMB * 2);
  unsigned short* wqt = (unsigned short*)alloc((size_t)QKV_N * EMB * 2);
  unsigned short* wpt = (unsigned short*)alloc((size_t)EMB * EMB * 2);
  unsigned short* Qb  = (unsigned short*)alloc((size_t)BATCH * NH * SEQ * HD * 2);
  unsigned short* Kb  = (unsigned short*)alloc((size_t)BATCH * NH * SEQ * HD * 2);
  unsigned short* Vt  = (unsigned short*)alloc((size_t)BATCH * NH * HD * SEQ * 2);
  unsigned short* Yb  = (unsigned short*)alloc((size_t)BATCH * SEQ * EMB * 2);

  const int nx = BATCH * SEQ * EMB;
  k_cvt_bf16<<<nx / 256, 256, 0, stream>>>(x, xb, nx);
  k_cvt_transpose<<<(QKV_N * EMB) / 256, 256, 0, stream>>>(Wqkv, wqt, EMB, QKV_N);
  k_cvt_transpose<<<(EMB * EMB) / 256, 256, 0, stream>>>(Wproj, wpt, EMB, EMB);

  k_gemm_qkv<<<dim3(QKV_N / 128, (BATCH * SEQ) / 128), 256, 0, stream>>>(
      xb, wqt, bqkv, Qb, Kb, Vt);

  k_attn<<<dim3(SEQ / 16, NH, BATCH), 32, 0, stream>>>(Qb, Kb, Vt, Yb);

  k_gemm_proj<<<dim3(EMB / 128, (BATCH * SEQ) / 128), 256, 0, stream>>>(
      Yb, wpt, bproj, out);
}
